// FourBitLinear_52982716564145
// MI455X (gfx1250) — matrix operand order repaired and checked by reading, not compiled, not run on hardware
//
#include <hip/hip_runtime.h>
#include <hip/hip_bf16.h>
#include <stdint.h>

#define IN_FEATURES 4096
#define OUT_FEATURES 11008
#define TOKENS 32

typedef __attribute__((ext_vector_type(16))) _Float16 v16h;
typedef __attribute__((ext_vector_type(8)))  _Float16 v8h;
typedef __attribute__((ext_vector_type(4)))  _Float16 v4h;
typedef __attribute__((ext_vector_type(2)))  _Float16 h2;
typedef __attribute__((ext_vector_type(8)))  float    v8f;
typedef __attribute__((ext_vector_type(4)))  float    v4f;
typedef __attribute__((ext_vector_type(4)))  int      v4i;

// ---------------------------------------------------------------------------
// Pass 1: convert x (f32) -> f16 copy in workspace (L2-resident, 256 KB) and
// compute per-token sums sx[t] = sum_i x[t,i] (needed for the zero-point term).
// One block per token.
// ---------------------------------------------------------------------------
__global__ __launch_bounds__(256) void fbl_prep_kernel(const float* __restrict__ x,
                                                       _Float16* __restrict__ xh,
                                                       float* __restrict__ sx) {
    __shared__ float red[256];
    const int m   = blockIdx.x;
    const int tid = threadIdx.x;
    const float*    xrow = x  + (size_t)m * IN_FEATURES;
    _Float16*       hrow = xh + (size_t)m * IN_FEATURES;
    float sum = 0.0f;
#pragma unroll
    for (int it = 0; it < IN_FEATURES / (256 * 4); ++it) {
        const int j = (it * 256 + tid) * 4;
        v4f v = *(const v4f*)(xrow + j);
        v4h h = { (_Float16)v.x, (_Float16)v.y, (_Float16)v.z, (_Float16)v.w };
        *(v4h*)(hrow + j) = h;
        sum += v.x + v.y + v.z + v.w;
    }
    red[tid] = sum;
    __syncthreads();
    for (int s = 128; s > 0; s >>= 1) {
        if (tid < s) red[tid] += red[tid + s];
        __syncthreads();
    }
    if (tid == 0) sx[m] = red[0];
}

// ---------------------------------------------------------------------------
// Pass 2: memory-bound dequant-GEMM.
//   Per wave: one N-tile of 16 outputs, both M-tiles (32 tokens), K = 4096.
//   B (weights): lane l -> row o0+(l&15); lanes 0-15 carry K 0..15,
//                lanes 16-31 carry K 16..31  => two global_load_b128 / lane.
//   Dequant trick: f16(0x6400 | n) == 1024+n exactly; pack two nibbles into a
//   dword and subtract (1032,1032) with one packed f16 add => (nib-8) exact.
//   A (x as f16, from L2): lane l -> row (l&15); K = sel*16 + {0..15}.
//   D = A x B + C via v_wmma_f32_16x16x32_f16, scale/zp/bias in the epilogue.
// ---------------------------------------------------------------------------
__global__ __launch_bounds__(256) void fbl_gemm_kernel(const int*   __restrict__ qw,
                                                       const float* __restrict__ scale,
                                                       const float* __restrict__ zp,
                                                       const float* __restrict__ bias,
                                                       const _Float16* __restrict__ xh,
                                                       const float* __restrict__ sx,
                                                       float* __restrict__ out) {
    const int lane = threadIdx.x & 31;
    const int wave = threadIdx.x >> 5;
    const int o0   = (blockIdx.x * 8 + wave) * 16;   // 16 outputs per wave
    const int nI   = lane & 15;
    const int sel  = lane >> 4;

    // Weight stream pointer: row stride = (IN/2) int32 = 8192 bytes.
    const char* qb  = (const char*)qw +
                      (size_t)(o0 + nI) * (size_t)(IN_FEATURES / 2 * 4) + sel * 32;
    // A-fragment pointers (f16 x copy in workspace).
    const char* pa0 = (const char*)xh + ((size_t)nI * IN_FEATURES + sel * 16) * 2;
    const char* pa1 = pa0 + (size_t)16 * IN_FEATURES * 2;

    v8f acc0 = {};
    v8f acc1 = {};
    const h2 kOff = { (_Float16)1032.0f, (_Float16)1032.0f };

#pragma unroll 2
    for (int kc = 0; kc < IN_FEATURES / 32; ++kc) {
        // 32 bytes of packed nibbles per lane (16 K-values).
        v4i bq0 = *(const v4i*)(qb);
        v4i bq1 = *(const v4i*)(qb + 16);
        __builtin_prefetch(qb + 1024, 0, 1);   // global_prefetch ~16 iters ahead

        // Dequantize: element 2j = high nibble (even K), 2j+1 = low nibble.
        union { v16h v; h2 p[8]; } B;
#pragma unroll
        for (int j = 0; j < 8; ++j) {
            const unsigned d = (unsigned)(j < 4 ? bq0[j] : bq1[j - 4]);
            const unsigned c = 0x64006400u | ((d >> 4) & 0xFu) | ((d & 0xFu) << 16);
            union { unsigned u; h2 h; } t;
            t.u = c;
            B.p[j] = t.h - kOff;               // exact (nib - 8) in f16
        }

        union { v16h v; v8h h[2]; } A0, A1;
        A0.h[0] = *(const v8h*)(pa0);
        A0.h[1] = *(const v8h*)(pa0 + 16);
        A1.h[0] = *(const v8h*)(pa1);
        A1.h[1] = *(const v8h*)(pa1 + 16);

        acc0 = __builtin_amdgcn_wmma_f32_16x16x32_f16(false, A0.v, false, B.v,
                                                      (short)0, acc0, false, false);
        acc1 = __builtin_amdgcn_wmma_f32_16x16x32_f16(false, A1.v, false, B.v,
                                                      (short)0, acc1, false, false);

        qb  += 64;   // 16 dwords = 32 K-values of nibbles
        pa0 += 64;   // 32 halves
        pa1 += 64;
    }

    // Epilogue: y = acc*scale/8 + sx[t]*zp + bias.
    const float s  = scale[o0 + nI] * 0.125f;
    const float z  = zp[o0 + nI];
    const float bb = bias[o0 + nI];
#pragma unroll
    for (int r = 0; r < 8; ++r) {
        const int t0 = sel * 8 + r;            // C/D layout: M = r + 8*sel
        out[(size_t)t0 * OUT_FEATURES + o0 + nI] = acc0[r] * s + sx[t0] * z + bb;
        const int t1 = 16 + t0;
        out[(size_t)t1 * OUT_FEATURES + o0 + nI] = acc1[r] * s + sx[t1] * z + bb;
    }
}

extern "C" void kernel_launch(void* const* d_in, const int* in_sizes, int n_in,
                              void* d_out, int out_size, void* d_ws, size_t ws_size,
                              hipStream_t stream) {
    (void)in_sizes; (void)n_in; (void)out_size; (void)ws_size;
    const float* x     = (const float*)d_in[0];
    const int*   qw    = (const int*)  d_in[1];
    const float* scale = (const float*)d_in[2];
    const float* zp    = (const float*)d_in[3];
    const float* bias  = (const float*)d_in[4];
    float*       out   = (float*)d_out;

    _Float16* xh = (_Float16*)d_ws;                                    // 256 KB
    float*    sx = (float*)((char*)d_ws + (size_t)TOKENS * IN_FEATURES * 2);

    fbl_prep_kernel<<<TOKENS, 256, 0, stream>>>(x, xh, sx);
    fbl_gemm_kernel<<<OUT_FEATURES / 128, 256, 0, stream>>>(qw, scale, zp, bias,
                                                            xh, sx, out);
}
